// Attention_Score_23407571763595
// MI455X (gfx1250) — matrix operand order repaired and checked by reading, not compiled, not run on hardware
//
#include <hip/hip_runtime.h>
#include <hip/hip_bf16.h>
#include <cstdint>

typedef __attribute__((ext_vector_type(16))) _Float16 v16h;
typedef __attribute__((ext_vector_type(8)))  _Float16 v8h;
typedef __attribute__((ext_vector_type(8)))  float    v8f;

#define DIM    768
#define NHEAD  12
#define HD     64
#define SEQ    1024
#define BATCH  8
#define BHND   (BATCH * NHEAD)     // 96
#define MROWS  (BATCH * SEQ)       // 8192
#define QKV_N  (3 * DIM)           // 2304
#define ATTN_SCALE 0.125f          // HD^-0.5
#define BPAD   40                  // padded LDS row stride (f16) for bank spread

// ---------------------------------------------------------------------------
// Fragment helpers (CDNA5 wave32 WMMA layouts, 16x16x32 f16 -> f32)
// A fragment (16x32, f16): lane holds row (lane&15); elems j at k = 16*hi + j
//   (hi = lane>>4)
// B fragment (32x16, f16): lane holds col (lane&15); elems j at k = 16*hi + j
// C/D (16x16, f32):        lane holds col (lane&15); VGPR r = row r + 8*hi
// ---------------------------------------------------------------------------

__device__ __forceinline__ v8f wmma_f16(v16h a, v16h b, v8f c) {
    return __builtin_amdgcn_wmma_f32_16x16x32_f16(
        false, a, false, b, (short)0, c, false, false);
}

__device__ __forceinline__ v16h load_afrag(const _Float16* __restrict__ base, int hi) {
    const v8h lo8 = *reinterpret_cast<const v8h*>(base + 16 * hi);
    const v8h hi8 = *reinterpret_cast<const v8h*>(base + 16 * hi + 8);
    v16h a;
#pragma unroll
    for (int i = 0; i < 8; ++i) { a[i] = lo8[i]; a[i + 8] = hi8[i]; }
    return a;
}

__device__ __forceinline__ v16h load_bfrag(const _Float16* __restrict__ base, int hi) {
    return *reinterpret_cast<const v16h*>(base + 16 * hi);
}

// 16-byte-aligned LDS read of a B fragment (two ds_load_b128)
__device__ __forceinline__ v16h load_bfrag_lds(const _Float16* base) {
    const v8h b0 = *reinterpret_cast<const v8h*>(base);
    const v8h b1 = *reinterpret_cast<const v8h*>(base + 8);
    v16h b;
#pragma unroll
    for (int i = 0; i < 8; ++i) { b[i] = b0[i]; b[i + 8] = b1[i]; }
    return b;
}

// ---------------------------------------------------------------------------
// gfx1250 async global->LDS copy (ASYNCcnt-tracked DMA path)
// ---------------------------------------------------------------------------
__device__ __forceinline__ void async_ld_b128(unsigned lds_off, const void* gaddr) {
    asm volatile("global_load_async_to_lds_b128 %0, %1, off"
                 :: "v"(lds_off), "v"((unsigned long long)(uintptr_t)gaddr)
                 : "memory");
}

__device__ __forceinline__ void wait_async0() {
#if __has_builtin(__builtin_amdgcn_s_wait_asynccnt)
    __builtin_amdgcn_s_wait_asynccnt(0);
#else
    asm volatile("s_wait_asynccnt 0x0" ::: "memory");
#endif
}

// ---------------------------------------------------------------------------
// Precision conversion kernels
// ---------------------------------------------------------------------------
__global__ void cvt_f32_to_f16_kernel(const float* __restrict__ in,
                                      _Float16* __restrict__ out, int n) {
    int i = blockIdx.x * blockDim.x + threadIdx.x;
    if (i < n) out[i] = (_Float16)in[i];
}

// in: [rows=K][cols=N] f32 row-major  ->  out: [N][K] f16 (transposed)
__global__ void transpose_cvt_kernel(const float* __restrict__ in,
                                     _Float16* __restrict__ out,
                                     int rows, int cols) {
    int i = blockIdx.x * blockDim.x + threadIdx.x;
    if (i < rows * cols) {
        int n = i / rows;
        int k = i - n * rows;
        out[i] = (_Float16)in[(size_t)k * cols + n];
    }
}

// ---------------------------------------------------------------------------
// QKV GEMM: block-cooperative 128x64 tile.  4 waves (32 rows each) share a
// B k-slab (64 cols x 32 k) staged in LDS by global_load_async_to_lds_b128,
// double-buffered (XOR address toggle, rolled loop so accumulators stay in
// fixed register tuples).  Qh, Kh per head [bh][n][d]; Vt transposed.
// ---------------------------------------------------------------------------
__global__ __launch_bounds__(128) void qkv_gemm_kernel(
    const _Float16* __restrict__ Xh, const _Float16* __restrict__ Wt,
    _Float16* __restrict__ Qh, _Float16* __restrict__ Kh,
    _Float16* __restrict__ Vt) {
    __shared__ __align__(16) _Float16 bsm[2][64][BPAD];

    const int tid  = threadIdx.x;
    const int lane = tid & 31;
    const int wave = tid >> 5;
    const int NT = QKV_N / 64;                 // 36
    const int bm = blockIdx.x / NT, nt = blockIdx.x % NT;
    const int m0 = bm * 128 + wave * 32, n0 = nt * 64;
    const int hi = lane >> 4, lo = lane & 15;

    const _Float16* __restrict__ arow0 = Xh + (size_t)(m0 + lo) * DIM;
    const _Float16* __restrict__ arow1 = arow0 + (size_t)16 * DIM;

    // fixed per-thread staging addresses (two 16B chunks per thread)
    const int row0 = tid >> 2,         part0 = tid & 3;
    const int row1 = (tid + 128) >> 2, part1 = tid & 3;
    const _Float16* gsrc0 = Wt + (size_t)(n0 + row0) * DIM + part0 * 8;
    const _Float16* gsrc1 = Wt + (size_t)(n0 + row1) * DIM + part1 * 8;
    const unsigned ldsA = (unsigned)(uintptr_t)&bsm[0][row0][part0 * 8];
    const unsigned ldsB = (unsigned)(uintptr_t)&bsm[0][row1][part1 * 8];
    const unsigned BUFB = 64 * BPAD * sizeof(_Float16);   // buffer stride bytes
    const unsigned BUFE = 64 * BPAD;                      // buffer stride elems

    // fixed per-lane LDS read base (B fragment origin)
    const _Float16* rbase = &bsm[0][lo][16 * hi];

    v8f acc[2][4];
#pragma unroll
    for (int mi = 0; mi < 2; ++mi)
#pragma unroll
        for (int t = 0; t < 4; ++t) acc[mi][t] = (v8f)0.f;

    // prologue: stage slab 0 into buffer 0
    async_ld_b128(ldsA, gsrc0);
    async_ld_b128(ldsB, gsrc1);
    wait_async0();
    __syncthreads();

    unsigned sbuf = BUFB;   // next stage target (buffer 1)
    unsigned roff = 0;      // current read buffer (element offset)
    const int NSLAB = DIM / 32;                // 24
#pragma clang loop unroll(disable)
    for (int ks = 0; ks < NSLAB; ++ks) {
        const int k = ks * 32;
        if (ks + 1 < NSLAB) {
            async_ld_b128(ldsA + sbuf, gsrc0 + k + 32);
            async_ld_b128(ldsB + sbuf, gsrc1 + k + 32);
        }

        __builtin_prefetch(arow0 + k + 128, 0, 3);
        __builtin_prefetch(arow1 + k + 128, 0, 3);
        const v16h a0 = load_afrag(arow0 + k, hi);
        const v16h a1 = load_afrag(arow1 + k, hi);
#pragma unroll
        for (int t = 0; t < 4; ++t) {
            const v16h b = load_bfrag_lds(rbase + roff + (size_t)(16 * t) * BPAD);
            acc[0][t] = wmma_f16(a0, b, acc[0][t]);
            acc[1][t] = wmma_f16(a1, b, acc[1][t]);
        }
        wait_async0();        // our slab k+1 DMA complete
        __syncthreads();      // all waves done reading cur + staged next
        sbuf ^= BUFB;
        roff ^= BUFE;
    }

    // decode output tile -> (s in {q,k,v}, head); 64-wide tile == whole head
    const int s  = n0 / DIM;
    const int h  = (n0 - s * DIM) / HD;
    const int b  = m0 / SEQ;
    const int n  = m0 - b * SEQ;
    const int bh = b * NHEAD + h;

#pragma unroll
    for (int mi = 0; mi < 2; ++mi) {
#pragma unroll
        for (int t = 0; t < 4; ++t) {
            const int d = 16 * t + lo;
#pragma unroll
            for (int r = 0; r < 8; ++r) {
                const int q = mi * 16 + r + 8 * hi;
                const _Float16 val = (_Float16)acc[mi][t][r];
                if (s == 0)
                    Qh[((size_t)bh * SEQ + n + q) * HD + d] = val;
                else if (s == 1)
                    Kh[((size_t)bh * SEQ + n + q) * HD + d] = val;
                else
                    Vt[((size_t)bh * HD + d) * SEQ + n + q] = val;
            }
        }
    }
}

// ---------------------------------------------------------------------------
// Flash-style attention: one wave per (bh, 32-row query block).  Two 16-row
// query tiles share the K and V fragments of each 32-wide kv block.
// Computes S^T = K * Q^T so the fp32 accumulator layout coincides with the
// A-fragment layout needed for P * V (no cross-lane transpose).
// ---------------------------------------------------------------------------
__global__ __launch_bounds__(128) void attn_kernel(
    const _Float16* __restrict__ Qh, const _Float16* __restrict__ Kh,
    const _Float16* __restrict__ Vt, _Float16* __restrict__ Oh) {
    const int lane = threadIdx.x & 31;
    const int wave = threadIdx.x >> 5;
    const int tile = blockIdx.x * 4 + wave;
    const int QT = SEQ / 32;                   // 32 query blocks
    const int bh = tile / QT;
    const int q0 = (tile - bh * QT) * 32;
    const int hi = lane >> 4, lo = lane & 15;

    const _Float16* __restrict__ qrow0 = Qh + ((size_t)bh * SEQ + q0 + lo) * HD;
    const _Float16* __restrict__ qrow1 = qrow0 + (size_t)16 * HD;
    v16h qb[2][2];
    qb[0][0] = load_bfrag(qrow0, hi);       qb[0][1] = load_bfrag(qrow0 + 32, hi);
    qb[1][0] = load_bfrag(qrow1, hi);       qb[1][1] = load_bfrag(qrow1 + 32, hi);

    float m[2]    = {-1e30f, -1e30f};
    float lsum[2] = {0.f, 0.f};
    v8f o[2][4];
#pragma unroll
    for (int qt = 0; qt < 2; ++qt)
#pragma unroll
        for (int t = 0; t < 4; ++t) o[qt][t] = (v8f)0.f;

    const _Float16* __restrict__ Kbh = Kh + (size_t)bh * SEQ * HD;
    const _Float16* __restrict__ Vbh = Vt + (size_t)bh * HD * SEQ;

    for (int kv = 0; kv < SEQ; kv += 32) {
        const _Float16* k0row = Kbh + (size_t)(kv + lo) * HD;
        const _Float16* k1row = Kbh + (size_t)(kv + 16 + lo) * HD;
        __builtin_prefetch(k0row + 2 * 32 * HD, 0, 3);
        __builtin_prefetch(k1row + 2 * 32 * HD, 0, 3);

        const v16h ka00 = load_afrag(k0row, hi);
        const v16h ka01 = load_afrag(k0row + 32, hi);
        const v16h ka10 = load_afrag(k1row, hi);
        const v16h ka11 = load_afrag(k1row + 32, hi);

        v16h pa[2];
#pragma unroll
        for (int qt = 0; qt < 2; ++qt) {
            v8f st0 = (v8f)0.f, st1 = (v8f)0.f;
            st0 = wmma_f16(ka00, qb[qt][0], st0);
            st0 = wmma_f16(ka01, qb[qt][1], st0);
            st1 = wmma_f16(ka10, qb[qt][0], st1);
            st1 = wmma_f16(ka11, qb[qt][1], st1);

            float s0[8], s1[8];
            float smax = -1e30f;
#pragma unroll
            for (int r = 0; r < 8; ++r) {
                s0[r] = st0[r] * ATTN_SCALE;
                s1[r] = st1[r] * ATTN_SCALE;
                smax = fmaxf(smax, fmaxf(s0[r], s1[r]));
            }
            smax = fmaxf(smax, __shfl_xor(smax, 16));
            const float mnew  = fmaxf(m[qt], smax);
            const float alpha = __expf(m[qt] - mnew);

            float rsum = 0.f;
#pragma unroll
            for (int r = 0; r < 8; ++r) {
                const float p0 = __expf(s0[r] - mnew);
                const float p1 = __expf(s1[r] - mnew);
                rsum += p0 + p1;
                pa[qt][r]     = (_Float16)p0;   // kv = 8*hi + r
                pa[qt][r + 8] = (_Float16)p1;   // kv = 16 + 8*hi + r
            }
            rsum += __shfl_xor(rsum, 16);
            lsum[qt] = lsum[qt] * alpha + rsum;
            m[qt] = mnew;

            float af[8];
#pragma unroll
            for (int r = 0; r < 8; ++r) af[r] = __shfl(alpha, r + 8 * hi);
#pragma unroll
            for (int r = 0; r < 8; ++r) {
                o[qt][0][r] *= af[r]; o[qt][1][r] *= af[r];
                o[qt][2][r] *= af[r]; o[qt][3][r] *= af[r];
            }
        }

        const _Float16* vb = Vbh + kv + 8 * hi;
#define VLD(p) __builtin_shufflevector(*reinterpret_cast<const v8h*>(p), *reinterpret_cast<const v8h*>((p) + 16), 0,1,2,3,4,5,6,7,8,9,10,11,12,13,14,15)
        const v16h v0 = VLD(vb + (size_t)(lo)      * SEQ);
        const v16h v1 = VLD(vb + (size_t)(16 + lo) * SEQ);
        const v16h v2 = VLD(vb + (size_t)(32 + lo) * SEQ);
        const v16h v3 = VLD(vb + (size_t)(48 + lo) * SEQ);
#undef VLD
#pragma unroll
        for (int qt = 0; qt < 2; ++qt) {
            o[qt][0] = wmma_f16(pa[qt], v0, o[qt][0]);
            o[qt][1] = wmma_f16(pa[qt], v1, o[qt][1]);
            o[qt][2] = wmma_f16(pa[qt], v2, o[qt][2]);
            o[qt][3] = wmma_f16(pa[qt], v3, o[qt][3]);
        }
    }

    const int b = bh / NHEAD, h = bh - (bh / NHEAD) * NHEAD;
#pragma unroll
    for (int qt = 0; qt < 2; ++qt) {
        const float linv = 1.0f / lsum[qt];
        float lf[8];
#pragma unroll
        for (int r = 0; r < 8; ++r) lf[r] = __shfl(linv, r + 8 * hi);

        _Float16* __restrict__ obase =
            Oh + (size_t)(b * SEQ + q0 + 16 * qt + 8 * hi) * DIM + h * HD + lo;
#pragma unroll
        for (int r = 0; r < 8; ++r) {
            obase[(size_t)r * DIM + 0]  = (_Float16)(o[qt][0][r] * lf[r]);
            obase[(size_t)r * DIM + 16] = (_Float16)(o[qt][1][r] * lf[r]);
            obase[(size_t)r * DIM + 32] = (_Float16)(o[qt][2][r] * lf[r]);
            obase[(size_t)r * DIM + 48] = (_Float16)(o[qt][3][r] * lf[r]);
        }
    }
}

// ---------------------------------------------------------------------------
// Projection GEMM: same block-cooperative 128x64 tiling with async-LDS
// staged weights; fp32 out + bias.
// ---------------------------------------------------------------------------
__global__ __launch_bounds__(128) void proj_gemm_kernel(
    const _Float16* __restrict__ Oh, const _Float16* __restrict__ Wt,
    const float* __restrict__ bias, float* __restrict__ out) {
    __shared__ __align__(16) _Float16 bsm[2][64][BPAD];

    const int tid  = threadIdx.x;
    const int lane = tid & 31;
    const int wave = tid >> 5;
    const int NT = DIM / 64;                   // 12
    const int bm = blockIdx.x / NT, nt = blockIdx.x % NT;
    const int m0 = bm * 128 + wave * 32, n0 = nt * 64;
    const int hi = lane >> 4, lo = lane & 15;

    const _Float16* __restrict__ arow0 = Oh + (size_t)(m0 + lo) * DIM;
    const _Float16* __restrict__ arow1 = arow0 + (size_t)16 * DIM;

    const int row0 = tid >> 2,         part0 = tid & 3;
    const int row1 = (tid + 128) >> 2, part1 = tid & 3;
    const _Float16* gsrc0 = Wt + (size_t)(n0 + row0) * DIM + part0 * 8;
    const _Float16* gsrc1 = Wt + (size_t)(n0 + row1) * DIM + part1 * 8;
    const unsigned ldsA = (unsigned)(uintptr_t)&bsm[0][row0][part0 * 8];
    const unsigned ldsB = (unsigned)(uintptr_t)&bsm[0][row1][part1 * 8];
    const unsigned BUFB = 64 * BPAD * sizeof(_Float16);
    const unsigned BUFE = 64 * BPAD;

    const _Float16* rbase = &bsm[0][lo][16 * hi];

    v8f acc[2][4];
#pragma unroll
    for (int mi = 0; mi < 2; ++mi)
#pragma unroll
        for (int t = 0; t < 4; ++t) acc[mi][t] = (v8f)0.f;

    async_ld_b128(ldsA, gsrc0);
    async_ld_b128(ldsB, gsrc1);
    wait_async0();
    __syncthreads();

    unsigned sbuf = BUFB;
    unsigned roff = 0;
    const int NSLAB = DIM / 32;
#pragma clang loop unroll(disable)
    for (int ks = 0; ks < NSLAB; ++ks) {
        const int k = ks * 32;
        if (ks + 1 < NSLAB) {
            async_ld_b128(ldsA + sbuf, gsrc0 + k + 32);
            async_ld_b128(ldsB + sbuf, gsrc1 + k + 32);
        }

        __builtin_prefetch(arow0 + k + 128, 0, 3);
        __builtin_prefetch(arow1 + k + 128, 0, 3);
        const v16h a0 = load_afrag(arow0 + k, hi);
        const v16h a1 = load_afrag(arow1 + k, hi);
#pragma unroll
        for (int t = 0; t < 4; ++t) {
            const v16h b = load_bfrag_lds(rbase + roff + (size_t)(16 * t) * BPAD);
            acc[0][t] = wmma_f16(a0, b, acc[0][t]);
            acc[1][t] = wmma_f16(a1, b, acc[1][t]);
        }
        wait_async0();
        __syncthreads();
        sbuf ^= BUFB;
        roff ^= BUFE;
    }

#pragma unroll
    for (int t = 0; t < 4; ++t) {
        const float bval = bias[n0 + 16 * t + lo];
#pragma unroll
        for (int mi = 0; mi < 2; ++mi)
#pragma unroll
            for (int r = 0; r < 8; ++r)
                out[(size_t)(m0 + mi * 16 + r + 8 * hi) * DIM + n0 + 16 * t + lo] =
                    acc[mi][t][r] + bval;
    }
}

// ---------------------------------------------------------------------------
// Host-side launch
// ---------------------------------------------------------------------------
extern "C" void kernel_launch(void* const* d_in, const int* in_sizes, int n_in,
                              void* d_out, int out_size, void* d_ws, size_t ws_size,
                              hipStream_t stream) {
    const float* x      = (const float*)d_in[0];
    const float* qkv_w  = (const float*)d_in[1];
    const float* proj_w = (const float*)d_in[2];
    const float* proj_b = (const float*)d_in[3];
    float* out = (float*)d_out;

    _Float16* ws = (_Float16*)d_ws;
    size_t off = 0;
    _Float16* Xh    = ws + off; off += (size_t)MROWS * DIM;
    _Float16* Wqkvt = ws + off; off += (size_t)DIM * QKV_N;
    _Float16* Wprjt = ws + off; off += (size_t)DIM * DIM;
    _Float16* Qh    = ws + off; off += (size_t)BHND * SEQ * HD;
    _Float16* Kh    = ws + off; off += (size_t)BHND * SEQ * HD;
    _Float16* Vt    = ws + off; off += (size_t)BHND * SEQ * HD;
    _Float16* Oh    = ws + off; off += (size_t)MROWS * DIM;

    const int nX = MROWS * DIM;
    cvt_f32_to_f16_kernel<<<(nX + 255) / 256, 256, 0, stream>>>(x, Xh, nX);

    const int nWq = DIM * QKV_N;
    transpose_cvt_kernel<<<(nWq + 255) / 256, 256, 0, stream>>>(qkv_w, Wqkvt, DIM, QKV_N);
    const int nWp = DIM * DIM;
    transpose_cvt_kernel<<<(nWp + 255) / 256, 256, 0, stream>>>(proj_w, Wprjt, DIM, DIM);

    // QKV GEMM: (8192/128) x (2304/64) = 64 x 36 blocks of 128 threads
    qkv_gemm_kernel<<<64 * 36, 128, 0, stream>>>(Xh, Wqkvt, Qh, Kh, Vt);

    // Attention: 96 bh * 32 q-blocks = 3072 waves
    attn_kernel<<<3072 / 4, 128, 0, stream>>>(Qh, Kh, Vt, Oh);

    // Projection: (8192/128) x (768/64) = 64 x 12 blocks
    proj_gemm_kernel<<<64 * 12, 128, 0, stream>>>(Oh, Wprjt, proj_b, out);
}
